// PointMixer_66675072303375
// MI455X (gfx1250) — matrix operand order repaired and checked by reading, not compiled, not run on hardware
//
#include <hip/hip_runtime.h>
#include <hip/hip_bf16.h>
#include <math.h>

// ---------------------------------------------------------------------------
// Types for CDNA5 WMMA (wave32, 16x16x32 f16 -> f32 accumulate)
// ---------------------------------------------------------------------------
typedef __attribute__((ext_vector_type(16))) _Float16 v16h;
typedef __attribute__((ext_vector_type(8)))  float    v8f;
typedef _Float16 half_t;

static constexpr int Bc  = 16;    // batch
static constexpr int NP  = 1024;  // points
static constexpr int KNB = 20;    // neighbors
static constexpr float EPSV = 1e-5f;

#define DEV __device__ __forceinline__

enum { ACT_NONE = 0, ACT_RELU = 1, ACT_LEAKY = 2, ACT_GELU = 3 };

DEV float actf(float v, int act) {
  if (act == ACT_RELU)  return v > 0.f ? v : 0.f;
  if (act == ACT_LEAKY) return v >= 0.f ? v : 0.2f * v;
  if (act == ACT_GELU)  return 0.5f * v * (1.f + erff(v * 0.70710678118654752f));
  return v;
}

DEV void atomicMaxF(float* addr, float val) {
  unsigned int* ua = (unsigned int*)addr;
  unsigned int old = *ua;
  while (true) {
    float oldf = __uint_as_float(old);
    if (oldf >= val) break;
    unsigned int assumed = old;
    old = atomicCAS(ua, assumed, __float_as_uint(val));
    if (old == assumed) break;
  }
}

// ---------------------------------------------------------------------------
// Generic WMMA GEMM: out[m,n] = act( sum_k A[m,k]*W[n,k] + bias[n] ) (+ resid)
// A: (M,K) f16 row-major, lda;  W: (N,K) f16 row-major, ldw (weights as (out,in))
// Optional batching over blockIdx.y (used for per-batch gram A @ A^T).
// ISA layouts (CDNA5 ISA):
//   A 16x32 f16: lane<16 -> row=lane, K={0..7,16..23}; lane>=16 -> K+=8
//   B 32x16 f16: lane = col, same K split as A (lane<16: K={0..7,16..23}; lane>=16: K+=8)
//   C/D f32: VGPR r -> (M = r + 8*(lane>=16), N = lane&15)
// ---------------------------------------------------------------------------
__global__ void gemm_f16(const half_t* __restrict__ A, int lda, long sA,
                         const half_t* __restrict__ W, int ldw, long sW,
                         const float* __restrict__ bias,
                         const float* __restrict__ resid,
                         float* __restrict__ out, int ldo, long sO,
                         half_t* __restrict__ outh,
                         int M, int N, int K, int act) {
  const int wave = threadIdx.x >> 5;
  const int lane = threadIdx.x & 31;
  const int tilesN = N >> 4;
  const long tile = (long)blockIdx.x * (blockDim.x >> 5) + wave;
  const long nTiles = (long)(M >> 4) * tilesN;
  if (tile >= nTiles) return;  // wave-uniform
  const int mT = (int)(tile / tilesN);
  const int nT = (int)(tile % tilesN);
  const int z = blockIdx.y;
  A += (long)z * sA;  W += (long)z * sW;  out += (long)z * sO;

  const int hi = lane >> 4;
  const int rl = lane & 15;
  const half_t* ap = A + (size_t)(mT * 16 + rl) * lda + hi * 8;
  const half_t* wp = W + (size_t)(nT * 16 + rl) * ldw + hi * 8;

  v8f c = {};
  for (int kc = 0; kc < K; kc += 32) {
    v16h a, b;
#pragma unroll
    for (int i = 0; i < 8; ++i) { a[i] = ap[kc + i]; a[8 + i] = ap[kc + 16 + i]; }
#pragma unroll
    for (int i = 0; i < 8; ++i) { b[i] = wp[kc + i]; b[8 + i] = wp[kc + 16 + i]; }
    c = __builtin_amdgcn_wmma_f32_16x16x32_f16(false, a, false, b, (short)0, c,
                                               false, false);
  }
  const int n = nT * 16 + rl;
  const float bs = bias ? bias[n] : 0.f;
#pragma unroll
  for (int r = 0; r < 8; ++r) {
    const int m = mT * 16 + hi * 8 + r;
    float v = actf(c[r] + bs, act);
    if (resid) v += resid[(size_t)m * ldo + n];
    out[(size_t)m * ldo + n] = v;
    if (outh) outh[(size_t)m * ldo + n] = (half_t)v;
  }
}

// ---------------------------------------------------------------------------
// Edge-conv WMMA, pass A: gather graph feature f=(x[nbr]-x[n] || x[n]) on the
// fly into the WMMA A operand; accumulate per-channel sum/sumsq for BN stats.
// M = B*NP*KNB rows, N = Cout, K = 2*Cin.
// ---------------------------------------------------------------------------
__global__ void edge_gemm_stats(const float* __restrict__ X, int ldx, int Cin,
                                const int* __restrict__ idx,
                                const half_t* __restrict__ W, int Cout,
                                float* __restrict__ sums, float* __restrict__ sqs) {
  __shared__ float s_sum[256], s_sq[256];
  for (int i = threadIdx.x; i < 256; i += blockDim.x) { s_sum[i] = 0.f; s_sq[i] = 0.f; }
  __syncthreads();

  const int wave = threadIdx.x >> 5, lane = threadIdx.x & 31;
  const int tilesN = Cout >> 4;
  const long M = (long)Bc * NP * KNB;
  const long tile = (long)blockIdx.x * (blockDim.x >> 5) + wave;
  const bool active = tile < (M >> 4) * tilesN;  // wave-uniform
  if (active) {
    const long mT = tile / tilesN;
    const int nT = (int)(tile % tilesN);
    const int hi = lane >> 4, rl = lane & 15;
    const long m = mT * 16 + rl;
    const int b = (int)(m / ((long)NP * KNB));
    const long rem = m % ((long)NP * KNB);
    const int n = (int)(rem / KNB), kk = (int)(rem % KNB);
    const int nbr = idx[((long)b * NP + n) * KNB + kk];
    const float* xn = X + ((long)b * NP + n) * ldx;
    const float* xb = X + ((long)b * NP + nbr) * ldx;
    const int K = 2 * Cin;
    const half_t* wp = W + (size_t)(nT * 16 + rl) * K + hi * 8;
    v8f c = {};
    for (int kc = 0; kc < K; kc += 32) {
      v16h a, bv;
#pragma unroll
      for (int i = 0; i < 16; ++i) {
        const int kl = ((i >> 3) << 4) + hi * 8 + (i & 7);
        const int cc = kc + kl;
        a[i] = (half_t)((cc < Cin) ? (xb[cc] - xn[cc]) : xn[cc - Cin]);
      }
#pragma unroll
      for (int i = 0; i < 8; ++i) { bv[i] = wp[kc + i]; bv[8 + i] = wp[kc + 16 + i]; }
      c = __builtin_amdgcn_wmma_f32_16x16x32_f16(false, a, false, bv, (short)0, c,
                                                 false, false);
    }
    const int ch = nT * 16 + rl;
    float ls = 0.f, lq = 0.f;
#pragma unroll
    for (int r = 0; r < 8; ++r) { ls += c[r]; lq += c[r] * c[r]; }
    atomicAdd(&s_sum[ch], ls);
    atomicAdd(&s_sq[ch], lq);
  }
  __syncthreads();
  for (int i = threadIdx.x; i < Cout; i += blockDim.x) {
    atomicAdd(&sums[i], s_sum[i]);
    atomicAdd(&sqs[i], s_sq[i]);
  }
}

// Pass B: recompute GEMM, apply BN + leaky, atomic-max over k into out slice.
__global__ void edge_gemm_apply(const float* __restrict__ X, int ldx, int Cin,
                                const int* __restrict__ idx,
                                const half_t* __restrict__ W, int Cout,
                                const float* __restrict__ sums,
                                const float* __restrict__ sqs,
                                const float* __restrict__ g,
                                const float* __restrict__ be,
                                float* __restrict__ out, int ldo) {
  const int wave = threadIdx.x >> 5, lane = threadIdx.x & 31;
  const int tilesN = Cout >> 4;
  const long M = (long)Bc * NP * KNB;
  const long tile = (long)blockIdx.x * (blockDim.x >> 5) + wave;
  if (tile >= (M >> 4) * tilesN) return;
  const long mT = tile / tilesN;
  const int nT = (int)(tile % tilesN);
  const int hi = lane >> 4, rl = lane & 15;
  const long m = mT * 16 + rl;
  const int b = (int)(m / ((long)NP * KNB));
  const long rem = m % ((long)NP * KNB);
  const int n = (int)(rem / KNB), kk = (int)(rem % KNB);
  const int nbr = idx[((long)b * NP + n) * KNB + kk];
  const float* xn = X + ((long)b * NP + n) * ldx;
  const float* xb = X + ((long)b * NP + nbr) * ldx;
  const int K = 2 * Cin;
  const half_t* wp = W + (size_t)(nT * 16 + rl) * K + hi * 8;
  v8f c = {};
  for (int kc = 0; kc < K; kc += 32) {
    v16h a, bv;
#pragma unroll
    for (int i = 0; i < 16; ++i) {
      const int kl = ((i >> 3) << 4) + hi * 8 + (i & 7);
      const int cc = kc + kl;
      a[i] = (half_t)((cc < Cin) ? (xb[cc] - xn[cc]) : xn[cc - Cin]);
    }
#pragma unroll
    for (int i = 0; i < 8; ++i) { bv[i] = wp[kc + i]; bv[8 + i] = wp[kc + 16 + i]; }
    c = __builtin_amdgcn_wmma_f32_16x16x32_f16(false, a, false, bv, (short)0, c,
                                               false, false);
  }
  const int ch = nT * 16 + rl;
  const float cnt = (float)M;
  const float mean = sums[ch] / cnt;
  const float inv = rsqrtf(sqs[ch] / cnt - mean * mean + EPSV);
  const float gg = g[ch], bb = be[ch];
#pragma unroll
  for (int r = 0; r < 8; ++r) {
    const long m2 = mT * 16 + hi * 8 + r;
    const int b2 = (int)(m2 / ((long)NP * KNB));
    const long r2 = m2 % ((long)NP * KNB);
    const int n2 = (int)(r2 / KNB);
    float v = (c[r] - mean) * inv * gg + bb;
    v = v >= 0.f ? v : 0.2f * v;
    atomicMaxF(&out[((long)b2 * NP + n2) * ldo + ch], v);
  }
}

// ---------------------------------------------------------------------------
// Stage-1 edge conv (Cin=3, feat=6, Cout=64) - tiny K, plain VALU.
// ---------------------------------------------------------------------------
__global__ void edge3_stats(const float* __restrict__ X, const int* __restrict__ idx,
                            const float* __restrict__ W,
                            float* __restrict__ sums, float* __restrict__ sqs) {
  __shared__ float s_sum[64], s_sq[64];
  if (threadIdx.x < 64) { s_sum[threadIdx.x] = 0.f; s_sq[threadIdx.x] = 0.f; }
  __syncthreads();
  const int co = threadIdx.x & 63;
  const int sub = threadIdx.x >> 6;
  const long m = (long)blockIdx.x * 4 + sub;
  const long Mtot = (long)Bc * NP * KNB;
  if (m < Mtot) {
    const int b = (int)(m / ((long)NP * KNB));
    const long rem = m % ((long)NP * KNB);
    const int n = (int)(rem / KNB), kk = (int)(rem % KNB);
    const int nbr = idx[((long)b * NP + n) * KNB + kk];
    const float* xn = X + ((long)b * NP + n) * 3;
    const float* xb = X + ((long)b * NP + nbr) * 3;
    const float f[6] = {xb[0] - xn[0], xb[1] - xn[1], xb[2] - xn[2],
                        xn[0], xn[1], xn[2]};
    const float* w = W + co * 6;
    float y = 0.f;
#pragma unroll
    for (int i = 0; i < 6; ++i) y += f[i] * w[i];
    atomicAdd(&s_sum[co], y);
    atomicAdd(&s_sq[co], y * y);
  }
  __syncthreads();
  if (threadIdx.x < 64) {
    atomicAdd(&sums[threadIdx.x], s_sum[threadIdx.x]);
    atomicAdd(&sqs[threadIdx.x], s_sq[threadIdx.x]);
  }
}

__global__ void edge3_apply(const float* __restrict__ X, const int* __restrict__ idx,
                            const float* __restrict__ W,
                            const float* __restrict__ sums, const float* __restrict__ sqs,
                            const float* __restrict__ g, const float* __restrict__ be,
                            float* __restrict__ out, int ldo) {
  const int co = threadIdx.x & 63;
  const int sub = threadIdx.x >> 6;
  const long m = (long)blockIdx.x * 4 + sub;
  const long Mtot = (long)Bc * NP * KNB;
  if (m >= Mtot) return;
  const int b = (int)(m / ((long)NP * KNB));
  const long rem = m % ((long)NP * KNB);
  const int n = (int)(rem / KNB), kk = (int)(rem % KNB);
  const int nbr = idx[((long)b * NP + n) * KNB + kk];
  const float* xn = X + ((long)b * NP + n) * 3;
  const float* xb = X + ((long)b * NP + nbr) * 3;
  const float f[6] = {xb[0] - xn[0], xb[1] - xn[1], xb[2] - xn[2],
                      xn[0], xn[1], xn[2]};
  const float* w = W + co * 6;
  float y = 0.f;
#pragma unroll
  for (int i = 0; i < 6; ++i) y += f[i] * w[i];
  const float cnt = (float)Mtot;
  const float mean = sums[co] / cnt;
  const float inv = rsqrtf(sqs[co] / cnt - mean * mean + EPSV);
  float v = (y - mean) * inv * g[co] + be[co];
  v = v >= 0.f ? v : 0.2f * v;
  atomicMaxF(&out[((long)b * NP + n) * ldo + co], v);
}

// ---------------------------------------------------------------------------
// kNN top-20: one wave per query row; score(m) = 2*dot[n,m] - |x_m|^2
// (the -|x_n|^2 row constant does not change ranking).
// ---------------------------------------------------------------------------
__global__ void topk_knn(const float* __restrict__ dot, const float* __restrict__ sq,
                         int* __restrict__ idx) {
  const long row = blockIdx.x;  // b*NP + n
  const int b = (int)(row / NP);
  const float* dr = dot + row * NP;
  const float* sqb = sq + (long)b * NP;
  const int lane = threadIdx.x & 31;
  unsigned mask = 0;
  for (int it = 0; it < KNB; ++it) {
    float bv = -3.4e38f;
    int bi = -1;
    for (int j = 0; j < 32; ++j) {
      if (mask & (1u << j)) continue;
      const int m = j * 32 + lane;
      const float s = 2.f * dr[m] - sqb[m];
      if (s > bv) { bv = s; bi = m; }
    }
    for (int off = 16; off; off >>= 1) {
      const float ov = __shfl_xor(bv, off, 32);
      const int oi = __shfl_xor(bi, off, 32);
      if (ov > bv || (ov == bv && oi >= 0 && oi < bi)) { bv = ov; bi = oi; }
    }
    if ((bi & 31) == lane) mask |= 1u << (bi >> 5);
    if (lane == 0) idx[row * KNB + it] = bi;
  }
}

// ---------------------------------------------------------------------------
// Small elementwise / reduction kernels
// ---------------------------------------------------------------------------
__global__ void fillk(float* p, float v, long n) {
  const long i = (long)blockIdx.x * blockDim.x + threadIdx.x;
  if (i < n) p[i] = v;
}

__global__ void f32_to_f16(const float* __restrict__ in, half_t* __restrict__ out, long n) {
  const long i = (long)blockIdx.x * blockDim.x + threadIdx.x;
  if (i < n) out[i] = (half_t)in[i];
}

__global__ void f32s_to_f16(const float* __restrict__ in, int ld, int C, long M,
                            half_t* __restrict__ out) {
  const long i = (long)blockIdx.x * blockDim.x + threadIdx.x;
  if (i >= M * C) return;
  const long m = i / C;
  const int c = (int)(i % C);
  out[i] = (half_t)in[m * ld + c];
}

// per-channel sum/sumsq of (M,C) matrix, one block per channel
__global__ void stats_rows(const float* __restrict__ X, int ld, long M,
                           float* __restrict__ sums, float* __restrict__ sqs) {
  const int ch = blockIdx.x;
  float s = 0.f, q = 0.f;
  for (long m = threadIdx.x; m < M; m += blockDim.x) {
    const float v = X[m * ld + ch];
    s += v; q += v * v;
  }
  __shared__ float ss[256], qq[256];
  ss[threadIdx.x] = s; qq[threadIdx.x] = q;
  __syncthreads();
  for (int o = 128; o; o >>= 1) {
    if ((int)threadIdx.x < o) { ss[threadIdx.x] += ss[threadIdx.x + o]; qq[threadIdx.x] += qq[threadIdx.x + o]; }
    __syncthreads();
  }
  if (threadIdx.x == 0) { sums[ch] = ss[0]; sqs[ch] = qq[0]; }
}

__global__ void bn_apply(float* __restrict__ X, int ld, int C, long M,
                         const float* __restrict__ sums, const float* __restrict__ sqs,
                         const float* __restrict__ g, const float* __restrict__ be,
                         int act, half_t* __restrict__ outh) {
  const long i = (long)blockIdx.x * blockDim.x + threadIdx.x;
  if (i >= M * C) return;
  const long m = i / C;
  const int ch = (int)(i % C);
  const float mean = sums[ch] / (float)M;
  const float inv = rsqrtf(sqs[ch] / (float)M - mean * mean + EPSV);
  float v = (X[m * ld + ch] - mean) * inv * g[ch] + be[ch];
  v = actf(v, act);
  X[m * ld + ch] = v;
  if (outh) outh[i] = (half_t)v;
}

__global__ void tpose3(const float* __restrict__ cloud, float* __restrict__ xt) {
  const long i = (long)blockIdx.x * blockDim.x + threadIdx.x;
  if (i >= (long)Bc * NP) return;
  const int b = (int)(i / NP), n = (int)(i % NP);
  const float* cb = cloud + (long)b * 3 * NP;
  xt[i * 3 + 0] = cb[n];
  xt[i * 3 + 1] = cb[NP + n];
  xt[i * 3 + 2] = cb[2 * NP + n];
}

__global__ void linear_k3(const float* __restrict__ X, const float* __restrict__ W,
                          const float* __restrict__ b, float* __restrict__ out) {
  const long i = (long)blockIdx.x * blockDim.x + threadIdx.x;
  if (i >= (long)Bc * NP * 64) return;
  const long m = i >> 6;
  const int co = (int)(i & 63);
  const float* x = X + m * 3;
  const float* w = W + co * 3;
  out[i] = x[0] * w[0] + x[1] * w[1] + x[2] * w[2] + b[co];
}

__global__ void linear_w6(const float* __restrict__ X, const float* __restrict__ W,
                          const float* __restrict__ b, float* __restrict__ out) {
  const long i = (long)blockIdx.x * blockDim.x + threadIdx.x;
  if (i >= (long)Bc * NP * 6) return;
  const long m = i / 6;
  const int co = (int)(i % 6);
  const float* x = X + m * 256;
  const float* w = W + co * 256;
  float acc = b[co];
  for (int k = 0; k < 256; ++k) acc += x[k] * w[k];
  out[i] = acc;
}

__global__ void colmax(const float* __restrict__ X, int C, float* __restrict__ out) {
  const long i = (long)blockIdx.x * blockDim.x + threadIdx.x;
  if (i >= (long)Bc * C) return;
  const int b = (int)(i / C), ch = (int)(i % C);
  float mx = -3.4e38f;
  for (int n = 0; n < NP; ++n) mx = fmaxf(mx, X[((long)b * NP + n) * C + ch]);
  out[i] = mx;
}

__global__ void fusion_build(const float* __restrict__ fa, const float* __restrict__ gb,
                             half_t* __restrict__ out) {
  const long i = (long)blockIdx.x * blockDim.x + threadIdx.x;
  if (i >= (long)Bc * NP * 2048) return;
  const long m = i >> 11;
  const int c = (int)(i & 2047);
  const float v = (c < 1024) ? fa[m * 1024 + c] : gb[(m / NP) * 1024 + (c - 1024)];
  out[i] = (half_t)v;
}

__global__ void rot_build(const float* __restrict__ pa, float* __restrict__ rt) {
  const int b = threadIdx.x;
  if (b >= Bc) return;
  const float az = pa[b * 6 + 0], ay = pa[b * 6 + 1], ax = pa[b * 6 + 2];
  const float cz = cosf(az), sz = sinf(az), cy = cosf(ay), sy = sinf(ay);
  const float cx = cosf(ax), sx = sinf(ax);
  float* R = rt + b * 12;
  R[0] = cz * cy;  R[1] = cz * sy * sx - sz * cx;  R[2] = cz * sy * cx + sz * sx;
  R[3] = sz * cy;  R[4] = sz * sy * sx + cz * cx;  R[5] = sz * sy * cx - cz * sx;
  R[6] = -sy;      R[7] = cy * sx;                 R[8] = cy * cx;
  R[9] = pa[b * 6 + 3]; R[10] = pa[b * 6 + 4]; R[11] = pa[b * 6 + 5];
}

__global__ void rot_apply(const float* __restrict__ cloud, const float* __restrict__ rt,
                          float* __restrict__ xt) {
  const long i = (long)blockIdx.x * blockDim.x + threadIdx.x;
  if (i >= (long)Bc * NP) return;
  const int b = (int)(i / NP), n = (int)(i % NP);
  const float* cb = cloud + (long)b * 3 * NP;
  const float x = cb[n], y = cb[NP + n], z = cb[2 * NP + n];
  const float* R = rt + b * 12;
  xt[i * 3 + 0] = R[0] * x + R[1] * y + R[2] * z + R[9];
  xt[i * 3 + 1] = R[3] * x + R[4] * y + R[5] * z + R[10];
  xt[i * 3 + 2] = R[6] * x + R[7] * y + R[8] * z + R[11];
}

__global__ void sqnorm(const float* __restrict__ X, int ld, int C, float* __restrict__ out) {
  const long i = (long)blockIdx.x * blockDim.x + threadIdx.x;
  if (i >= (long)Bc * NP) return;
  const float* x = X + i * ld;
  float s = 0.f;
  for (int c = 0; c < C; ++c) s += x[c] * x[c];
  out[i] = s;
}

__global__ void gram3(const float* __restrict__ X, float* __restrict__ dot) {
  const long i = (long)blockIdx.x * blockDim.x + threadIdx.x;
  if (i >= (long)Bc * NP * NP) return;
  const long b = i / ((long)NP * NP);
  const long rem = i % ((long)NP * NP);
  const int n = (int)(rem / NP), m = (int)(rem % NP);
  const float* xn = X + (b * NP + n) * 3;
  const float* xm = X + (b * NP + m) * 3;
  dot[i] = xn[0] * xm[0] + xn[1] * xm[1] + xn[2] * xm[2];
}

// y (B*NP,128) -> x (B,128,NP)
__global__ void y_to_x(const float* __restrict__ y, float* __restrict__ x) {
  const long i = (long)blockIdx.x * blockDim.x + threadIdx.x;
  if (i >= (long)Bc * 128 * NP) return;
  const long b = i / (128L * NP);
  const long rem = i % (128L * NP);
  const int c = (int)(rem / NP), n = (int)(rem % NP);
  x[i] = y[((long)b * NP + n) * 128 + c];
}

// x[b,c,n] += t2[(b*NP+n)*128 + c]
__global__ void resid_tadd(float* __restrict__ x, const float* __restrict__ t2) {
  const long i = (long)blockIdx.x * blockDim.x + threadIdx.x;
  if (i >= (long)Bc * 128 * NP) return;
  const long b = i / (128L * NP);
  const long rem = i % (128L * NP);
  const int c = (int)(rem / NP), n = (int)(rem % NP);
  x[i] += t2[((long)b * NP + n) * 128 + c];
}

// token LN: per (b,n) normalize across 128 channels; write f16 A[(b*128+c)*NP+n]
__global__ void ln_token(const float* __restrict__ x, const float* __restrict__ g,
                         const float* __restrict__ be, half_t* __restrict__ A) {
  const long i = (long)blockIdx.x * blockDim.x + threadIdx.x;
  if (i >= (long)Bc * NP) return;
  const int b = (int)(i / NP), n = (int)(i % NP);
  const float* xp = x + (long)b * 128 * NP + n;
  float s = 0.f;
  for (int c = 0; c < 128; ++c) s += xp[(long)c * NP];
  const float mean = s / 128.f;
  float q = 0.f;
  for (int c = 0; c < 128; ++c) { const float d = xp[(long)c * NP] - mean; q += d * d; }
  const float inv = rsqrtf(q / 128.f + EPSV);
  for (int c = 0; c < 128; ++c)
    A[((long)b * 128 + c) * NP + n] =
        (half_t)((xp[(long)c * NP] - mean) * inv * g[c] + be[c]);
}

// channel LN: per (b,c) row normalize across NP; write f16 T[(b*NP+n)*128+c]
__global__ void ln_chan(const float* __restrict__ x, const float* __restrict__ g,
                        const float* __restrict__ be, half_t* __restrict__ T) {
  const long row = blockIdx.x;  // b*128 + c
  const int b = (int)(row >> 7), c = (int)(row & 127);
  const float* xp = x + row * NP;
  __shared__ float ss[256], qq[256];
  float s = 0.f, q = 0.f;
  for (int n = threadIdx.x; n < NP; n += 256) { const float v = xp[n]; s += v; q += v * v; }
  ss[threadIdx.x] = s; qq[threadIdx.x] = q;
  __syncthreads();
  for (int o = 128; o; o >>= 1) {
    if ((int)threadIdx.x < o) { ss[threadIdx.x] += ss[threadIdx.x + o]; qq[threadIdx.x] += qq[threadIdx.x + o]; }
    __syncthreads();
  }
  const float mean = ss[0] / (float)NP;
  const float inv = rsqrtf(qq[0] / (float)NP - mean * mean + EPSV);
  for (int n = threadIdx.x; n < NP; n += 256) {
    const float v = (xp[n] - mean) * inv * g[n] + be[n];
    T[((long)b * NP + n) * 128 + c] = (half_t)v;
  }
}

// ---------------------------------------------------------------------------
// Host orchestration
// ---------------------------------------------------------------------------
extern "C" void kernel_launch(void* const* d_in, const int* in_sizes, int n_in,
                              void* d_out, int out_size, void* d_ws, size_t ws_size,
                              hipStream_t stream) {
  (void)in_sizes; (void)n_in; (void)out_size; (void)ws_size;
#define INF(i) ((const float*)d_in[i])
  const float* cloud_a = INF(0);
  const float* cloud_b = INF(1);
  // tnet params (setup_inputs dict insertion order)
  const float *w1 = INF(2), *b1 = INF(3), *tg1 = INF(4), *tbe1 = INF(5);
  const float *w2 = INF(6), *b2 = INF(7), *tg2 = INF(8), *tbe2 = INF(9);
  const float *w3 = INF(10), *b3 = INF(11), *tg3 = INF(12), *tbe3 = INF(13);
  const float *w4 = INF(14), *b4 = INF(15), *w5 = INF(16), *b5 = INF(17);
  const float *w6 = INF(18), *b6 = INF(19);
  // gcnn
  const float *gc1 = INF(20), *gg1 = INF(21), *gbe1 = INF(22);
  const float *gc2 = INF(23), *gg2 = INF(24), *gbe2 = INF(25);
  const float *gc3 = INF(26), *gg3 = INF(27), *gbe3 = INF(28);
  const float *gc4 = INF(29), *gg4 = INF(30), *gbe4 = INF(31);
  const float *gc5 = INF(32), *gg5 = INF(33), *gbe5 = INF(34);
  const int MIX1 = 35, MIX2 = 47;  // 12 params each: ln_t_g,ln_t_b,t_w1,t_b1,t_w2,t_b2,ln_c_g,ln_c_b,c_w1,c_b1,c_w2,c_b2

  const long ROWS = (long)Bc * NP;  // 16384
  const size_t MB = 1u << 20;

  char* ws = (char*)d_ws;
  size_t cur = 0;
  auto alloc = [&](size_t bytes) -> char* {
    char* p = ws + cur;
    cur += (bytes + 255) & ~(size_t)255;
    return p;
  };
  // weight f16 copies
  half_t* w2h = (half_t*)alloc(128 * 64 * 2);
  half_t* w3h = (half_t*)alloc(1024 * 128 * 2);
  half_t* w4h = (half_t*)alloc(512 * 2048 * 2);
  half_t* w5h = (half_t*)alloc(256 * 512 * 2);
  half_t* c2h = (half_t*)alloc(64 * 128 * 2);
  half_t* c3h = (half_t*)alloc(128 * 128 * 2);
  half_t* c4h = (half_t*)alloc(256 * 256 * 2);
  half_t* c5h = (half_t*)alloc(128 * 512 * 2);
  half_t* mtw1h[2], *mtw2h[2], *mcw1h[2], *mcw2h[2];
  for (int i = 0; i < 2; ++i) {
    mtw1h[i] = (half_t*)alloc(1024 * 1024 * 2);
    mtw2h[i] = (half_t*)alloc(1024 * 1024 * 2);
    mcw1h[i] = (half_t*)alloc(128 * 128 * 2);
    mcw2h[i] = (half_t*)alloc(128 * 128 * 2);
  }
  // small runtime buffers
  float* XT = (float*)alloc(ROWS * 3 * 4);
  float* H1 = (float*)alloc(ROWS * 64 * 4);
  half_t* H1H = (half_t*)alloc(ROWS * 64 * 2);
  float* H2 = (float*)alloc(ROWS * 128 * 4);
  half_t* H2H = (half_t*)alloc(ROWS * 128 * 2);
  float* SUM = (float*)alloc(1024 * 4);
  float* SQS = (float*)alloc(1024 * 4);
  float* GB = (float*)alloc(Bc * 1024 * 4);
  float* H6 = (float*)alloc(ROWS * 6 * 4);
  float* PA = (float*)alloc(Bc * 6 * 4);
  float* RT = (float*)alloc(Bc * 12 * 4);
  float* SQ = (float*)alloc(ROWS * 4);
  int* IDX = (int*)alloc(ROWS * KNB * 4);
  // big reusable regions
  char* big0 = alloc(64 * MB);  // fa/fb (tnet)  <->  DOT gram (gcnn)
  char* big1 = alloc(64 * MB);  // FUSH (tnet)   <->  CAT | XTH | Y (gcnn)
  char* big2 = alloc(64 * MB);  // H4 + H4H      <->  AT | HT | HTH (mixer)
  char* big3 = alloc(32 * MB);  // H5 + H5H      <->  TC | HC | HCH | T2 (mixer)

  float* BIG0 = (float*)big0;
  half_t* FUSH = (half_t*)big1;
  float* CAT = (float*)big1;
  half_t* XTH = (half_t*)(big1 + 32 * MB);
  float* Y = (float*)(big1 + 48 * MB);
  float* H4 = (float*)big2;
  half_t* H4H = (half_t*)(big2 + 32 * MB);
  half_t* AT = (half_t*)big2;
  float* HT = (float*)(big2 + 4 * MB);
  half_t* HTH = (half_t*)(big2 + 12 * MB);
  float* H5 = (float*)big3;
  half_t* H5H = (half_t*)(big3 + 16 * MB);
  half_t* TC = (half_t*)big3;
  float* HC = (float*)(big3 + 4 * MB);
  half_t* HCH = (half_t*)(big3 + 12 * MB);
  float* T2 = (float*)(big3 + 16 * MB);

  float* xa = (float*)d_out;
  float* xb = xa + (long)Bc * 128 * NP;

  auto grd = [](long n) { return dim3((unsigned)((n + 255) / 256)); };
  auto cvt = [&](const float* s, half_t* d, long n) {
    f32_to_f16<<<grd(n), 256, 0, stream>>>(s, d, n);
  };
  auto gemm = [&](const half_t* A, int lda, long sA, const half_t* W, int ldw, long sW,
                  const float* bias, const float* resid, float* out, int ldo, long sO,
                  half_t* outh, int M, int N, int K, int act, int batch) {
    long tiles = (long)(M / 16) * (N / 16);
    dim3 g((unsigned)((tiles + 7) / 8), (unsigned)batch, 1);
    gemm_f16<<<g, 256, 0, stream>>>(A, lda, sA, W, ldw, sW, bias, resid, out, ldo, sO,
                                    outh, M, N, K, act);
  };

  // ---- convert all GEMM weights to f16 -------------------------------------
  cvt(w2, w2h, 128 * 64);       cvt(w3, w3h, 1024 * 128);
  cvt(w4, w4h, 512 * 2048);     cvt(w5, w5h, 256 * 512);
  cvt(gc2, c2h, 64 * 128);      cvt(gc3, c3h, 128 * 128);
  cvt(gc4, c4h, 256 * 256);     cvt(gc5, c5h, 128 * 512);
  for (int i = 0; i < 2; ++i) {
    const int pb = i ? MIX2 : MIX1;
    cvt(INF(pb + 2), mtw1h[i], 1024 * 1024);
    cvt(INF(pb + 4), mtw2h[i], 1024 * 1024);
    cvt(INF(pb + 8), mcw1h[i], 128 * 128);
    cvt(INF(pb + 10), mcw2h[i], 128 * 128);
  }

  // ---- tnet shared MLP1: 3->64->128->1024 with BN+ReLU ---------------------
  auto mlp1 = [&](const float* cloud) {  // result -> BIG0 (ROWS,1024)
    tpose3<<<grd(ROWS), 256, 0, stream>>>(cloud, XT);
    linear_k3<<<grd(ROWS * 64), 256, 0, stream>>>(XT, w1, b1, H1);
    stats_rows<<<64, 256, 0, stream>>>(H1, 64, ROWS, SUM, SQS);
    bn_apply<<<grd(ROWS * 64), 256, 0, stream>>>(H1, 64, 64, ROWS, SUM, SQS, tg1, tbe1,
                                                 ACT_RELU, H1H);
    gemm(H1H, 64, 0, w2h, 64, 0, b2, nullptr, H2, 128, 0, nullptr, (int)ROWS, 128, 64,
         ACT_NONE, 1);
    stats_rows<<<128, 256, 0, stream>>>(H2, 128, ROWS, SUM, SQS);
    bn_apply<<<grd(ROWS * 128), 256, 0, stream>>>(H2, 128, 128, ROWS, SUM, SQS, tg2,
                                                  tbe2, ACT_RELU, H2H);
    gemm(H2H, 128, 0, w3h, 128, 0, b3, nullptr, BIG0, 1024, 0, nullptr, (int)ROWS, 1024,
         128, ACT_NONE, 1);
    stats_rows<<<1024, 256, 0, stream>>>(BIG0, 1024, ROWS, SUM, SQS);
    bn_apply<<<grd(ROWS * 1024), 256, 0, stream>>>(BIG0, 1024, 1024, ROWS, SUM, SQS,
                                                   tg3, tbe3, ACT_RELU, nullptr);
  };

  mlp1(cloud_b);                                               // fb -> BIG0
  colmax<<<grd(Bc * 1024), 256, 0, stream>>>(BIG0, 1024, GB);  // gb
  mlp1(cloud_a);                                               // fa -> BIG0
  fusion_build<<<grd(ROWS * 2048), 256, 0, stream>>>(BIG0, GB, FUSH);
  gemm(FUSH, 2048, 0, w4h, 2048, 0, b4, nullptr, H4, 512, 0, H4H, (int)ROWS, 512, 2048,
       ACT_RELU, 1);
  gemm(H4H, 512, 0, w5h, 512, 0, b5, nullptr, H5, 256, 0, H5H, (int)ROWS, 256, 512,
       ACT_RELU, 1);
  linear_w6<<<grd(ROWS * 6), 256, 0, stream>>>(H5, w6, b6, H6);
  colmax<<<1, 256, 0, stream>>>(H6, 6, PA);  // (Bc*6 = 96 threads needed)
  rot_build<<<1, 32, 0, stream>>>(PA, RT);
  rot_apply<<<grd(ROWS), 256, 0, stream>>>(cloud_a, RT, XT);  // ca_t as (B,NP,3)

  // ---- DGCNN ---------------------------------------------------------------
  auto edge_stage = [&](const float* Xin, int ldx, int Cin, const half_t* Wh, int Cout,
                        const float* g, const float* be, float* outBase) {
    f32s_to_f16<<<grd(ROWS * Cin), 256, 0, stream>>>(Xin, ldx, Cin, ROWS, XTH);
    sqnorm<<<grd(ROWS), 256, 0, stream>>>(Xin, ldx, Cin, SQ);
    gemm(XTH, Cin, (long)NP * Cin, XTH, Cin, (long)NP * Cin, nullptr, nullptr, BIG0, NP,
         (long)NP * NP, nullptr, NP, NP, Cin, ACT_NONE, Bc);  // per-batch gram
    topk_knn<<<(unsigned)ROWS, 32, 0, stream>>>(BIG0, SQ, IDX);
    fillk<<<grd(1024), 256, 0, stream>>>(SUM, 0.f, 1024);
    fillk<<<grd(1024), 256, 0, stream>>>(SQS, 0.f, 1024);
    const long tiles = (ROWS * KNB / 16) * (Cout / 16);
    edge_gemm_stats<<<(unsigned)((tiles + 7) / 8), 256, 0, stream>>>(Xin, ldx, Cin, IDX,
                                                                     Wh, Cout, SUM, SQS);
    edge_gemm_apply<<<(unsigned)((tiles + 7) / 8), 256, 0, stream>>>(
        Xin, ldx, Cin, IDX, Wh, Cout, SUM, SQS, g, be, outBase, 512);
  };

  auto gcnn = [&](float* xout) {  // input xt in XT (B,NP,3)
    fillk<<<grd(ROWS * 512), 256, 0, stream>>>(CAT, -3.4e38f, ROWS * 512);
    // stage1 (Cin=3): plain
    sqnorm<<<grd(ROWS), 256, 0, stream>>>(XT, 3, 3, SQ);
    gram3<<<grd((long)Bc * NP * NP), 256, 0, stream>>>(XT, BIG0);
    topk_knn<<<(unsigned)ROWS, 32, 0, stream>>>(BIG0, SQ, IDX);
    fillk<<<grd(1024), 256, 0, stream>>>(SUM, 0.f, 1024);
    fillk<<<grd(1024), 256, 0, stream>>>(SQS, 0.f, 1024);
    const long M1 = ROWS * KNB;
    edge3_stats<<<(unsigned)((M1 + 3) / 4), 256, 0, stream>>>(XT, IDX, gc1, SUM, SQS);
    edge3_apply<<<(unsigned)((M1 + 3) / 4), 256, 0, stream>>>(XT, IDX, gc1, SUM, SQS,
                                                              gg1, gbe1, CAT, 512);
    // stages 2-4 (WMMA edge conv)
    edge_stage(CAT + 0, 512, 64, c2h, 64, gg2, gbe2, CAT + 64);
    edge_stage(CAT + 64, 512, 64, c3h, 128, gg3, gbe3, CAT + 128);
    edge_stage(CAT + 128, 512, 128, c4h, 256, gg4, gbe4, CAT + 256);
    // c5: (ROWS,512) @ (128,512)^T, BN + leaky
    f32s_to_f16<<<grd(ROWS * 512), 256, 0, stream>>>(CAT, 512, 512, ROWS, XTH);
    gemm(XTH, 512, 0, c5h, 512, 0, nullptr, nullptr, Y, 128, 0, nullptr, (int)ROWS, 128,
         512, ACT_NONE, 1);
    stats_rows<<<128, 256, 0, stream>>>(Y, 128, ROWS, SUM, SQS);
    bn_apply<<<grd(ROWS * 128), 256, 0, stream>>>(Y, 128, 128, ROWS, SUM, SQS, gg5,
                                                  gbe5, ACT_LEAKY, nullptr);
    y_to_x<<<grd((long)Bc * 128 * NP), 256, 0, stream>>>(Y, xout);
  };

  // ---- mixer ---------------------------------------------------------------
  auto mixer = [&](float* x, int pb, int wi) {
    ln_token<<<grd(ROWS), 256, 0, stream>>>(x, INF(pb + 0), INF(pb + 1), AT);
    gemm(AT, NP, 0, mtw1h[wi], NP, 0, INF(pb + 3), nullptr, HT, NP, 0, HTH, Bc * 128,
         NP, NP, ACT_GELU, 1);
    gemm(HTH, NP, 0, mtw2h[wi], NP, 0, INF(pb + 5), x, x, NP, 0, nullptr, Bc * 128, NP,
         NP, ACT_NONE, 1);
    ln_chan<<<Bc * 128, 256, 0, stream>>>(x, INF(pb + 6), INF(pb + 7), TC);
    gemm(TC, 128, 0, mcw1h[wi], 128, 0, INF(pb + 9), nullptr, HC, 128, 0, HCH,
         (int)ROWS, 128, 128, ACT_GELU, 1);
    gemm(HCH, 128, 0, mcw2h[wi], 128, 0, INF(pb + 11), nullptr, T2, 128, 0, nullptr,
         (int)ROWS, 128, 128, ACT_NONE, 1);
    resid_tadd<<<grd((long)Bc * 128 * NP), 256, 0, stream>>>(x, T2);
  };

  // ---- full pipeline -------------------------------------------------------
  gcnn(xa);                                   // ga from ca_t (already in XT)
  mixer(xa, MIX1, 0);
  mixer(xa, MIX2, 1);

  tpose3<<<grd(ROWS), 256, 0, stream>>>(cloud_b, XT);
  gcnn(xb);                                   // gb from cloud_b
  mixer(xb, MIX1, 0);
  mixer(xb, MIX2, 1);
#undef INF
}
